// CTCDecoder_31576599561112
// MI455X (gfx1250) — compile-verified, matrix operand order checked by reading, not run on hardware
//
#include <hip/hip_runtime.h>
#include <hip/hip_bf16.h>

// ---------------------------------------------------------------------------
// CDNA5 (gfx1250) bidirectional-GRU stack + vocab projection.
// Matmuls via v_wmma_f32_16x16x32_bf16 (fp32 accumulate); per-step activation
// blocks staged into LDS with the Tensor Data Mover (tensor_load_to_lds,
// TENSORcnt double-buffering).
// ---------------------------------------------------------------------------

typedef __bf16 bf16;
typedef __attribute__((ext_vector_type(16))) __bf16 v16bf;
typedef __attribute__((ext_vector_type(8)))  float  v8f;
typedef __attribute__((ext_vector_type(4))) unsigned int uint32x4;
typedef __attribute__((ext_vector_type(8))) int          int32x8;
typedef __attribute__((ext_vector_type(4))) int          int32x4;

struct Frag32B { uint4 a, b; };   // 32 bytes == one v16bf fragment

__device__ __forceinline__ bf16 f2bf(float x) {
    unsigned u = __builtin_bit_cast(unsigned, x);
    unsigned r = u + 0x7FFFu + ((u >> 16) & 1u);     // round-to-nearest-even
    unsigned short h = (unsigned short)(r >> 16);
    return __builtin_bit_cast(bf16, h);
}
__device__ __forceinline__ float bf2f(bf16 b) {
    unsigned short s = __builtin_bit_cast(unsigned short, b);
    return __builtin_bit_cast(float, ((unsigned)s) << 16);
}

// Load one 16x32 bf16 A/B fragment for v_wmma_f32_16x16x32_bf16.
// Caller passes p = base + row*ld + ktile + (lane>>4)*8 ; layout per ISA:
//   elements 0..7  -> K = k'..k'+7,  elements 8..15 -> K = k'+16..k'+23
__device__ __forceinline__ v16bf ld_frag(const bf16* p) {
    Frag32B f;
    f.a = *(const uint4*)(p);
    f.b = *(const uint4*)(p + 16);
    return __builtin_bit_cast(v16bf, f);
}

__device__ __forceinline__ v8f wmma_bf16(v16bf a, v16bf b, v8f c) {
    return __builtin_amdgcn_wmma_f32_16x16x32_bf16(
        /*neg_a=*/false, a, /*neg_b=*/false, b,
        /*c_mod=*/(short)0, c, /*reuse_a=*/false, /*reuse_b=*/false);
}

__device__ __forceinline__ v8f vzero() {
    v8f z;
#pragma unroll
    for (int e = 0; e < 8; ++e) z[e] = 0.f;
    return z;
}

// ---------------------------------------------------------------------------
// TDM: 1-D contiguous fp32 tile (nelem elements, nelem <= 65535) -> LDS.
// D# packing per CDNA5 ISA 8.3/8.4:
//   g0: [1:0]=count=1, [63:32]=lds_addr, [120:64]=global_addr, [127:126]=2
//   g1: [17:16]=data_size(2 -> 4B), [79:48]=tensor_dim0, [111:80]=tensor_dim1,
//       [127:112]=tile_dim0, [143:128]=tile_dim1(0), [207:160]=dim0_stride
// ---------------------------------------------------------------------------
__device__ __forceinline__ void tdm_load_1d_f32(unsigned lds_off, const float* gsrc,
                                                unsigned nelem) {
    unsigned long long ga = (unsigned long long)(const void*)gsrc;
    uint32x4 g0;
    g0[0] = 1u;                                              // count=1, user-mode
    g0[1] = lds_off;                                         // lds_addr (bytes)
    g0[2] = (unsigned)(ga & 0xFFFFFFFFu);                    // global_addr[31:0]
    g0[3] = (unsigned)((ga >> 32) & 0x01FFFFFFu) | (2u << 30); // addr[56:32] | type=2
    int32x8 g1;
    g1[0] = (int)(2u << 16);                                 // data_size = 4 bytes
    g1[1] = (int)((nelem & 0xFFFFu) << 16);                  // tensor_dim0[15:0]
    g1[2] = (int)(((nelem >> 16) & 0xFFFFu) | (1u << 16));   // dim0[31:16] | dim1=1
    g1[3] = (int)((nelem & 0xFFFFu) << 16);                  // dim1 hi=0 | tile_dim0
    g1[4] = 0;                                               // tile_dim1=0, tile_dim2=0
    g1[5] = (int)nelem;                                      // dim0_stride[31:0]
    g1[6] = 0;
    g1[7] = 0;
    int32x4 gz4;
    gz4[0] = 0; gz4[1] = 0; gz4[2] = 0; gz4[3] = 0;
    int32x8 gz8;
#pragma unroll
    for (int i = 0; i < 8; ++i) gz8[i] = 0;
    // clang-23 / therock-10.0 6-arg form
    __builtin_amdgcn_tensor_load_to_lds(g0, g1, gz4, gz4, gz8, 0);
}

// Problem constants
#define BB 32
#define TT 512
#define FF 512
#define UU 256
#define VV 6000
#define MTOT (BB * TT)   // 16384 rows, time-major: row = t*B + b

// ---------------------------------------------------------------------------
// Weight transpose + fp32 -> bf16:  src[k*rs + n]  ->  dst[n*K + k]
// ---------------------------------------------------------------------------
__global__ void k_transpose_cvt(const float* __restrict__ src, bf16* __restrict__ dst,
                                int K, int N, int rs) {
    int total = K * N;
    for (int idx = blockIdx.x * blockDim.x + threadIdx.x; idx < total;
         idx += gridDim.x * blockDim.x) {
        int k = idx / N, n = idx - k * N;
        dst[(size_t)n * K + k] = f2bf(src[(size_t)k * rs + n]);
    }
}

// x[b][t][f] (fp32) -> x_bf[t][b][f] (bf16, time-major)
__global__ void k_x_cvt(const float* __restrict__ x, bf16* __restrict__ dst) {
    size_t total = (size_t)BB * TT * FF;
    for (size_t idx = (size_t)blockIdx.x * blockDim.x + threadIdx.x; idx < total;
         idx += (size_t)gridDim.x * blockDim.x) {
        int f = (int)(idx & (FF - 1));
        size_t bt = idx >> 9;
        int t = (int)(bt & (TT - 1));
        int b = (int)(bt >> 9);
        dst[(((size_t)t * BB + b) << 9) + f] = f2bf(x[idx]);
    }
}

// ---------------------------------------------------------------------------
// C[M,N] = A[M,K](bf16 row-major) * Bt[N,K](bf16, weightT) + bias[N]
// Block = 256 threads = 8 waves; wave tile 64x32 (4x2 sub-tiles of 16x16),
// block tile 128x128. permute!=0 remaps time-major row t*B+b -> b*T+t.
// ---------------------------------------------------------------------------
__launch_bounds__(256)
__global__ void k_gemm_bias(const bf16* __restrict__ A, const bf16* __restrict__ Bt,
                            const float* __restrict__ bias, float* __restrict__ C,
                            int M, int N, int K, int permute) {
    const int lane = threadIdx.x & 31;
    const int w    = threadIdx.x >> 5;
    const int wm   = w >> 2;          // 0..1
    const int wn   = w & 3;           // 0..3
    const int m0   = blockIdx.x * 128 + wm * 64;
    const int n0   = blockIdx.y * 128 + wn * 32;
    const int hi   = lane >> 4;
    const int lm   = lane & 15;

    v8f acc[4][2];
#pragma unroll
    for (int i = 0; i < 4; ++i)
#pragma unroll
        for (int j = 0; j < 2; ++j) acc[i][j] = vzero();

    for (int k = 0; k < K; k += 32) {
        if (k + 32 < K)  // gfx1250 global_prefetch_b8 for next K tile
            __builtin_prefetch(A + (size_t)(m0 + lm) * K + k + 32, 0, 3);
        v16bf a[4], b[2];
#pragma unroll
        for (int i = 0; i < 4; ++i)
            a[i] = ld_frag(A + (size_t)(m0 + i * 16 + lm) * K + k + hi * 8);
#pragma unroll
        for (int j = 0; j < 2; ++j) {
            int nr = n0 + j * 16 + lm;
            if (nr > N - 1) nr = N - 1;                 // clamp (tiles either fully valid or skipped)
            b[j] = ld_frag(Bt + (size_t)nr * K + k + hi * 8);
        }
#pragma unroll
        for (int i = 0; i < 4; ++i)
#pragma unroll
            for (int j = 0; j < 2; ++j)
                acc[i][j] = wmma_bf16(a[i], b[j], acc[i][j]);
    }

#pragma unroll
    for (int j = 0; j < 2; ++j) {
        const int ncol = n0 + j * 16 + lm;
        if (ncol >= N) continue;
        const float bv = bias ? bias[ncol] : 0.f;
#pragma unroll
        for (int i = 0; i < 4; ++i) {
#pragma unroll
            for (int r = 0; r < 8; ++r) {
                const int m = m0 + i * 16 + r + hi * 8;   // D layout: VGPR r -> M = r + 8*hi
                size_t row = (size_t)m;
                if (permute) {                            // t*B+b -> b*T+t
                    int b_ = m & (BB - 1);
                    int t_ = m >> 5;
                    row = (size_t)b_ * TT + t_;
                }
                C[row * (size_t)N + ncol] = acc[i][j][r] + bv;
            }
        }
    }
}

// ---------------------------------------------------------------------------
// Persistent GRU recurrence: one workgroup per direction (grid=2).
// h kept in LDS (bf16); per-step Gx/Cx blocks prefetched into LDS by the
// Tensor Data Mover, double-buffered, overlapped with phase-1 WMMA work.
// Dynamic LDS layout (245760 B total):
//   gxs : 2 x [32*512] f32  (131072 B)
//   cxs : 2 x [32*256] f32  ( 65536 B)
//   h_bf/rh_bf/u_bf : 3 x [32*256] bf16 (49152 B)
// ---------------------------------------------------------------------------
#define GRU_LDS_BYTES (2 * BB * 512 * 4 + 2 * BB * UU * 4 + 3 * BB * UU * 2)

__launch_bounds__(256)
__global__ void k_gru(const bf16* __restrict__ Wgh,   // [dir][512][256]
                      const bf16* __restrict__ Wch,   // [dir][256][256]
                      const float* __restrict__ Gx,   // [dir][T*B][512] (bias included)
                      const float* __restrict__ Cx,   // [dir][T*B][256] (bias included)
                      bf16* __restrict__ feats_bf,    // [T][B][2U] next-layer input
                      float* __restrict__ dout)       // null, or feats out [B][T][2U]
{
    const int d    = blockIdx.x;
    const int lane = threadIdx.x & 31;
    const int w    = threadIdx.x >> 5;
    const int hi   = lane >> 4;
    const int lm   = lane & 15;

    const bf16*  wgh = Wgh + (size_t)d * 512 * 256;
    const bf16*  wch = Wch + (size_t)d * 256 * 256;
    const float* gx  = Gx  + (size_t)d * MTOT * 512;
    const float* cx  = Cx  + (size_t)d * MTOT * 256;

    extern __shared__ char smem[];
    float* gxs  = (float*)smem;                 // [2][32*512]
    float* cxs  = gxs + 2 * BB * 512;           // [2][32*256]
    bf16*  h_bf = (bf16*)(cxs + 2 * BB * UU);   // [32*256]
    bf16*  rh_bf = h_bf + BB * UU;
    bf16*  u_bf  = rh_bf + BB * UU;

    // LDS byte offsets (generic-pointer low 32 bits == LDS offset)
    const unsigned gxs_off = (unsigned)(size_t)(void*)gxs;
    const unsigned cxs_off = (unsigned)(size_t)(void*)cxs;

    // Prologue: TDM-stage step 0 activations into buffer 0, init h = 0.
    const int t0 = d ? (TT - 1) : 0;
    if (w == 0) {
        tdm_load_1d_f32(gxs_off, gx + (size_t)t0 * BB * 512, BB * 512);
        tdm_load_1d_f32(cxs_off, cx + (size_t)t0 * BB * UU, BB * UU);
        __builtin_amdgcn_s_wait_tensorcnt(0);
    }
    for (int i = threadIdx.x; i < BB * UU; i += 256) h_bf[i] = f2bf(0.f);
    __syncthreads();

    int buf = 0;
    for (int step = 0; step < TT; ++step) {
        const int t = d ? (TT - 1 - step) : step;
        const float* gxl = gxs + buf * BB * 512;
        const float* cxl = cxs + buf * BB * UU;

        // ---- phase 1: RU = h @ WghT (M=32, N=512, K=256); wave w owns n in [w*64, w*64+64)
        {
            v8f acc[2][4];
#pragma unroll
            for (int i = 0; i < 2; ++i)
#pragma unroll
                for (int j = 0; j < 4; ++j) acc[i][j] = vzero();
            const int n0 = w * 64;
            for (int k = 0; k < 256; k += 32) {
                v16bf a[2], b[4];
#pragma unroll
                for (int i = 0; i < 2; ++i)
                    a[i] = ld_frag(&h_bf[(i * 16 + lm) * UU + k + hi * 8]);
#pragma unroll
                for (int j = 0; j < 4; ++j)
                    b[j] = ld_frag(wgh + (size_t)(n0 + j * 16 + lm) * 256 + k + hi * 8);
#pragma unroll
                for (int i = 0; i < 2; ++i)
#pragma unroll
                    for (int j = 0; j < 4; ++j)
                        acc[i][j] = wmma_bf16(a[i], b[j], acc[i][j]);
            }

            // Overlap: TDM-prefetch next step's activations into the other buffer.
            if (w == 0 && step + 1 < TT) {
                const int tn = d ? (TT - 2 - step) : (step + 1);
                const unsigned bo = (buf ^ 1);
                tdm_load_1d_f32(gxs_off + bo * BB * 512 * 4,
                                gx + (size_t)tn * BB * 512, BB * 512);
                tdm_load_1d_f32(cxs_off + bo * BB * UU * 4,
                                cx + (size_t)tn * BB * UU, BB * UU);
            }

#pragma unroll
            for (int j = 0; j < 4; ++j) {
                const int n = n0 + j * 16 + lm;
#pragma unroll
                for (int i = 0; i < 2; ++i) {
#pragma unroll
                    for (int r = 0; r < 8; ++r) {
                        const int m = i * 16 + r + 8 * hi;
                        float g = acc[i][j][r] + gxl[m * 512 + n];
                        float s = 1.f / (1.f + __expf(-g));
                        if (n < UU) {           // reset gate -> r*h
                            rh_bf[m * UU + n] = f2bf(s * bf2f(h_bf[m * UU + n]));
                        } else {                // update gate
                            u_bf[m * UU + (n - UU)] = f2bf(s);
                        }
                    }
                }
            }
        }
        __syncthreads();

        // ---- phase 2: Ct = (r*h) @ WchT (M=32, N=256, K=256); wave w owns n in [w*32, w*32+32)
        {
            v8f acc[2][2];
#pragma unroll
            for (int i = 0; i < 2; ++i)
#pragma unroll
                for (int j = 0; j < 2; ++j) acc[i][j] = vzero();
            const int n0 = w * 32;
            for (int k = 0; k < 256; k += 32) {
                v16bf a[2], b[2];
#pragma unroll
                for (int i = 0; i < 2; ++i)
                    a[i] = ld_frag(&rh_bf[(i * 16 + lm) * UU + k + hi * 8]);
#pragma unroll
                for (int j = 0; j < 2; ++j)
                    b[j] = ld_frag(wch + (size_t)(n0 + j * 16 + lm) * 256 + k + hi * 8);
#pragma unroll
                for (int i = 0; i < 2; ++i)
#pragma unroll
                    for (int j = 0; j < 2; ++j)
                        acc[i][j] = wmma_bf16(a[i], b[j], acc[i][j]);
            }
#pragma unroll
            for (int j = 0; j < 2; ++j) {
                const int n = n0 + j * 16 + lm;
#pragma unroll
                for (int i = 0; i < 2; ++i) {
#pragma unroll
                    for (int r = 0; r < 8; ++r) {
                        const int m = i * 16 + r + 8 * hi;
                        float c  = tanhf(acc[i][j][r] + cxl[m * UU + n]);
                        float u  = bf2f(u_bf[m * UU + n]);
                        float h  = bf2f(h_bf[m * UU + n]);
                        float hn = u * h + (1.f - u) * c;
                        h_bf[m * UU + n] = f2bf(hn);
                        feats_bf[((size_t)t * BB + m) * (2 * UU) + d * UU + n] = f2bf(hn);
                        if (dout)
                            dout[((size_t)m * TT + t) * (2 * UU) + d * UU + n] = hn;
                    }
                }
            }
        }
        // Fence the in-flight TDM prefetch before anyone reads the other buffer.
        if (w == 0) __builtin_amdgcn_s_wait_tensorcnt(0);
        __syncthreads();
        buf ^= 1;
    }
}

// ---------------------------------------------------------------------------
extern "C" void kernel_launch(void* const* d_in, const int* in_sizes, int n_in,
                              void* d_out, int out_size, void* d_ws, size_t ws_size,
                              hipStream_t stream) {
    const float* x    = (const float*)d_in[0];
    const float* Wg   = (const float*)d_in[1];   // [2][2][768][512]
    const float* bg   = (const float*)d_in[2];   // [2][2][512]
    const float* Wc   = (const float*)d_in[3];   // [2][2][768][256]
    const float* bc   = (const float*)d_in[4];   // [2][2][256]
    const float* Wout = (const float*)d_in[5];   // [512][6000]
    const float* bout = (const float*)d_in[6];   // [6000]
    (void)in_sizes; (void)n_in; (void)out_size; (void)ws_size;

    char* ws = (char*)d_ws;
    size_t off = 0;
    auto alloc = [&](size_t bytes) -> void* {
        off = (off + 255) & ~(size_t)255;
        void* p = ws + off;
        off += bytes;
        return p;
    };

    bf16* Wgx_bf = (bf16*)alloc((size_t)4 * 512 * 512 * sizeof(bf16)); // [l*2+d][N=512][K=512]
    bf16* Wgh_bf = (bf16*)alloc((size_t)4 * 512 * 256 * sizeof(bf16)); // [l*2+d][N=512][K=256]
    bf16* Wcx_bf = (bf16*)alloc((size_t)4 * 256 * 512 * sizeof(bf16)); // [l*2+d][N=256][K=512]
    bf16* Wch_bf = (bf16*)alloc((size_t)4 * 256 * 256 * sizeof(bf16)); // [l*2+d][N=256][K=256]
    bf16* Wout_bf= (bf16*)alloc((size_t)VV * 512 * sizeof(bf16));      // [N=6000][K=512]
    bf16* x_bf   = (bf16*)alloc((size_t)MTOT * 512 * sizeof(bf16));    // layer-0 input (time-major)
    bf16* f1_bf  = (bf16*)alloc((size_t)MTOT * 512 * sizeof(bf16));    // layer-0 output
    bf16* f2_bf  = (bf16*)alloc((size_t)MTOT * 512 * sizeof(bf16));    // layer-1 output
    float* Gx    = (float*)alloc((size_t)2 * MTOT * 512 * sizeof(float)); // per-layer gate pre-acts
    float* Cx    = (float*)alloc((size_t)2 * MTOT * 256 * sizeof(float)); // per-layer cand pre-acts

    // ---- one-time weight conversion (deterministic; redone each call) ----
    for (int ld_ = 0; ld_ < 4; ++ld_) {
        k_transpose_cvt<<<256, 256, 0, stream>>>(Wg + (size_t)ld_ * 768 * 512,
                                                 Wgx_bf + (size_t)ld_ * 512 * 512, 512, 512, 512);
        k_transpose_cvt<<<128, 256, 0, stream>>>(Wg + ((size_t)ld_ * 768 + 512) * 512,
                                                 Wgh_bf + (size_t)ld_ * 512 * 256, 256, 512, 512);
        k_transpose_cvt<<<128, 256, 0, stream>>>(Wc + (size_t)ld_ * 768 * 256,
                                                 Wcx_bf + (size_t)ld_ * 256 * 512, 512, 256, 256);
        k_transpose_cvt<<<64, 256, 0, stream>>>(Wc + ((size_t)ld_ * 768 + 512) * 256,
                                                Wch_bf + (size_t)ld_ * 256 * 256, 256, 256, 256);
    }
    k_transpose_cvt<<<2048, 256, 0, stream>>>(Wout, Wout_bf, 512, VV, VV);
    k_x_cvt<<<4096, 256, 0, stream>>>(x, x_bf);

    // ---- layers ----
    const bf16* lin[2]  = { x_bf, f1_bf };
    bf16*       lout[2] = { f1_bf, f2_bf };
    for (int l = 0; l < 2; ++l) {
        for (int d = 0; d < 2; ++d) {
            const int ld_ = l * 2 + d;
            // Gx = feats @ Wg_x + bg  : [16384,512] x [512->512]
            k_gemm_bias<<<dim3(128, 4), 256, 0, stream>>>(
                lin[l], Wgx_bf + (size_t)ld_ * 512 * 512, bg + (size_t)ld_ * 512,
                Gx + (size_t)d * MTOT * 512, MTOT, 512, 512, 0);
            // Cx = feats @ Wc_x + bc  : [16384,512] x [512->256]
            k_gemm_bias<<<dim3(128, 2), 256, 0, stream>>>(
                lin[l], Wcx_bf + (size_t)ld_ * 256 * 512, bc + (size_t)ld_ * 256,
                Cx + (size_t)d * MTOT * 256, MTOT, 256, 512, 0);
        }
        // recurrence over T (fw + bw in parallel workgroups), TDM-staged activations
        k_gru<<<2, 256, GRU_LDS_BYTES, stream>>>(
            Wgh_bf + (size_t)l * 2 * 512 * 256, Wch_bf + (size_t)l * 2 * 256 * 256,
            Gx, Cx, lout[l], (l == 1) ? (float*)d_out : nullptr);
    }

    // ---- final projection: [16384,512] x [512->6000] + bout, rows remapped t*B+b -> b*T+t
    float* final_out = (float*)d_out + (size_t)BB * TT * (2 * UU);
    k_gemm_bias<<<dim3(128, 47), 256, 0, stream>>>(
        f2_bf, Wout_bf, bout, final_out, MTOT, VV, 512, 1);
}
